// EdgeSAGELayer_8701603742217
// MI455X (gfx1250) — matrix operand order repaired and checked by reading, not compiled, not run on hardware
//
#include <hip/hip_runtime.h>

// Problem constants (from reference)
#define N_NODES 100000
#define N_EDGES 600000
#define EDGE_FEAT 128
#define IN_CH 512
#define OUT_CH 512
#define K_DIM (IN_CH + EDGE_FEAT) // 640

// GEMM tiling
#define BM 128         // rows per block (8 waves x 16)
#define BN 256         // cols per block
#define KC 128         // K chunk staged in LDS: BN*KC*2 = 64 KB per buffer
#define NKC (K_DIM / KC) // 5 chunks

typedef __bf16 bf16_t;
typedef __attribute__((ext_vector_type(16))) bf16_t v16bf;
typedef __attribute__((ext_vector_type(8)))  bf16_t v8bf;
typedef __attribute__((ext_vector_type(8)))  float  v8f;
typedef __attribute__((ext_vector_type(4)))  float  v4f;

__device__ __forceinline__ unsigned short f32_to_bf16_rne(float f) {
    unsigned u = __builtin_bit_cast(unsigned, f);
    unsigned r = u + 0x7FFFu + ((u >> 16) & 1u);
    return (unsigned short)(r >> 16);
}

// ---------------------------------------------------------------- zero ws
__global__ void zero_f32_kernel(float* __restrict__ p, int n) {
    int i = blockIdx.x * 256 + threadIdx.x;
    if (i < n) p[i] = 0.0f;
}

// ------------------------------------------------- convert W [512,640] -> bf16
__global__ void convw_kernel(const float* __restrict__ W, unsigned short* __restrict__ Wb, int n) {
    int i = blockIdx.x * 256 + threadIdx.x;
    if (i < n) Wb[i] = f32_to_bf16_rne(W[i]);
}

// ------------------------------------------------- scatter-sum over edges
// thread -> (edge, 4-feature chunk); 32 threads per edge.
// edge_attr is streamed once -> non-temporal loads (don't pollute L2).
__global__ void scatter_kernel(const float* __restrict__ edge_attr,
                               const long long* __restrict__ edge_index, // row 0 = target
                               float* __restrict__ sums,
                               float* __restrict__ counts) {
    int t = blockIdx.x * 256 + threadIdx.x;
    int e = t >> 5;
    int c = t & 31;
    if (e >= N_EDGES) return;
    int tgt = (int)edge_index[e]; // edge_index[0][e]
    v4f v = __builtin_nontemporal_load(
        (const v4f*)(edge_attr + (size_t)e * EDGE_FEAT + c * 4));
    float* dst = sums + (size_t)tgt * EDGE_FEAT + c * 4;
    atomicAdd(dst + 0, v.x);
    atomicAdd(dst + 1, v.y);
    atomicAdd(dst + 2, v.z);
    atomicAdd(dst + 3, v.w);
    if (c == 0) atomicAdd(counts + tgt, 1.0f);
}

// ------------------------------------------------- pack h = [node_attr | mean] as bf16
// thread -> 8 consecutive k of one row (16B store). N_NODES * 80 threads.
// Sources are read once -> NT loads. The h stores stay regular-temporal on
// purpose: h (128 MB) fits in the 192 MB L2 and is re-read by the GEMM.
__global__ void pack_h_kernel(const float* __restrict__ node_attr,
                              const float* __restrict__ sums,
                              const float* __restrict__ counts,
                              unsigned short* __restrict__ h) {
    int t = blockIdx.x * 256 + threadIdx.x;
    if (t >= N_NODES * (K_DIM / 8)) return;
    int n = t / (K_DIM / 8);
    int k = (t % (K_DIM / 8)) * 8;
    v4f f0, f1;
    if (k < IN_CH) {
        const float* p = node_attr + (size_t)n * IN_CH + k;
        f0 = __builtin_nontemporal_load((const v4f*)(p));
        f1 = __builtin_nontemporal_load((const v4f*)(p + 4));
    } else {
        float inv = 1.0f / fmaxf(counts[n], 1.0f);
        const float* p = sums + (size_t)n * EDGE_FEAT + (k - IN_CH);
        f0 = __builtin_nontemporal_load((const v4f*)(p));
        f1 = __builtin_nontemporal_load((const v4f*)(p + 4));
        f0 *= inv;
        f1 *= inv;
    }
    union { unsigned short s[8]; uint4 u; } r;
    r.s[0] = f32_to_bf16_rne(f0.x); r.s[1] = f32_to_bf16_rne(f0.y);
    r.s[2] = f32_to_bf16_rne(f0.z); r.s[3] = f32_to_bf16_rne(f0.w);
    r.s[4] = f32_to_bf16_rne(f1.x); r.s[5] = f32_to_bf16_rne(f1.y);
    r.s[6] = f32_to_bf16_rne(f1.z); r.s[7] = f32_to_bf16_rne(f1.w);
    *(uint4*)(h + (size_t)n * K_DIM + k) = r.u;
}

// ------------------------------------------------- bf16 WMMA GEMM + bias + sigmoid
// Grid: (2 col-blocks, 782 row-blocks) -- col-blocks sharing the same h rows
// dispatch adjacently so the second one L2-hits (h fits in the 192 MB L2).
// Block: 256 threads = 8 waves; tile 128 rows x 256 cols; 16 acc per wave.
// W staged via GLOBAL_LOAD_ASYNC_TO_LDS_B128 into double-buffered LDS.

__device__ __forceinline__ void stage_chunk_async(unsigned short* ldsbuf,
                                                  const unsigned short* Wb,
                                                  int colBase, int kc, int tid) {
    #pragma unroll
    for (int i = 0; i < (BN * KC) / (256 * 8); ++i) { // 16 x b128 per thread
        int elem = (i * 256 + tid) * 8;   // ushort index within chunk
        int c = elem >> 7;                // / KC
        int k = elem & (KC - 1);
        unsigned ldsoff = (unsigned)(uintptr_t)(ldsbuf + elem); // LDS byte offset
        const unsigned short* g = Wb + (size_t)(colBase + c) * K_DIM + kc * KC + k;
        asm volatile("global_load_async_to_lds_b128 %0, %1, off"
                     :: "v"(ldsoff), "v"(g)
                     : "memory");
    }
}

__device__ __forceinline__ void wait_async_all() {
    asm volatile("s_wait_asynccnt 0x0" ::: "memory");
}

__launch_bounds__(256)
__global__ void gemm_kernel(const unsigned short* __restrict__ h,  // bf16 bits [N][640]
                            const unsigned short* __restrict__ Wb, // bf16 bits [512][640]
                            const float* __restrict__ bias,
                            float* __restrict__ out) {
    extern __shared__ unsigned short ldsW[]; // 2 * BN*KC ushorts = 128 KB

    const int tid = threadIdx.x;
    const int wave = tid >> 5;
    const int lane = tid & 31;
    const int colBase = blockIdx.x * BN;               // x = column tile (2)
    const int rowBase = blockIdx.y * BM + wave * 16;   // y = row tile (782)

    // A-fragment addressing (contiguous k order k = 16h + s, same order as B):
    // lanes 0-15: row M=lane, K {0..15}; lanes 16-31: row M=lane-16, K {16..31}
    int arow = rowBase + (lane & 15);
    if (arow > N_NODES - 1) arow = N_NODES - 1; // clamp: keeps EXEC all-ones for WMMA
    const int koff = (lane >> 4) * 16;
    const unsigned short* hrow = h + (size_t)arow * K_DIM;

    // B-fragment addressing: lane column = lane&15, half-wave selects K 0..15 / 16..31
    const int bcol = lane & 15;
    const int bksel = (lane >> 4) * 16;

    v8f acc[16];
    #pragma unroll
    for (int ct = 0; ct < 16; ++ct) acc[ct] = {};

    // Prologue: fill buffer 0 with chunk 0
    stage_chunk_async(ldsW, Wb, colBase, 0, tid);
    wait_async_all();
    __syncthreads();

    for (int kc = 0; kc < NKC; ++kc) {
        // Issue async fill of the next chunk into the other buffer
        if (kc + 1 < NKC) {
            stage_chunk_async(ldsW + ((kc + 1) & 1) * (BN * KC), Wb, colBase, kc + 1, tid);
            __builtin_prefetch(hrow + (kc + 1) * KC + koff, 0, 0);
        }

        const unsigned short* buf = ldsW + (kc & 1) * (BN * KC);
        const unsigned short* bbase = buf + (size_t)bcol * KC + bksel;

        #pragma unroll
        for (int kt = 0; kt < KC / 32; ++kt) { // 4 K-steps of 32
            const int kk = kc * KC + kt * 32 + koff;
            v8bf lo = *(const v8bf*)(hrow + kk);
            v8bf hi = *(const v8bf*)(hrow + kk + 8);
            v16bf a = __builtin_shufflevector(lo, hi,
                0, 1, 2, 3, 4, 5, 6, 7, 8, 9, 10, 11, 12, 13, 14, 15);

            // Paired B fragments: both live before the first WMMA of the pair,
            // so the allocator must keep two 8-VGPR buffers -> one s_wait_dscnt
            // per two WMMAs, and each load pair hides under two matrix ops.
            const unsigned short* bk = bbase + kt * 32;
            #pragma unroll
            for (int ct = 0; ct < 16; ct += 2) {
                v16bf b0 = *(const v16bf*)(bk + (size_t)(ct + 0) * 16 * KC);
                v16bf b1 = *(const v16bf*)(bk + (size_t)(ct + 1) * 16 * KC);
                acc[ct + 0] = __builtin_amdgcn_wmma_f32_16x16x32_bf16(
                    false, a, false, b0, (short)0, acc[ct + 0], false, false);
                acc[ct + 1] = __builtin_amdgcn_wmma_f32_16x16x32_bf16(
                    false, a, false, b1, (short)0, acc[ct + 1], false, false);
            }
        }

        // My async fills (issued at loop top) must be done, then workgroup-wide
        // barrier publishes the next buffer and releases the one just read.
        wait_async_all();
        __syncthreads();
    }

    // Epilogue (C layout: lanes 0-15 -> N=lane, M=vgpr; lanes 16-31 -> N=lane-16, M=vgpr+8)
    // out is written once -> non-temporal stores keep h/W resident in L2.
    const int crow0 = rowBase + (lane >> 4) * 8;
    #pragma unroll
    for (int ct = 0; ct < 16; ++ct) {
        int col = colBase + ct * 16 + (lane & 15);
        float bv = bias[col];
        #pragma unroll
        for (int i = 0; i < 8; ++i) {
            int r = crow0 + i;
            if (r < N_NODES) {
                float x = acc[ct][i] + bv;
                float y = 1.0f / (1.0f + __expf(-x));
                __builtin_nontemporal_store(y, &out[(size_t)r * OUT_CH + col]);
            }
        }
    }
}

extern "C" void kernel_launch(void* const* d_in, const int* in_sizes, int n_in,
                              void* d_out, int out_size, void* d_ws, size_t ws_size,
                              hipStream_t stream) {
    const float*     edge_attr  = (const float*)d_in[0];
    const long long* edge_index = (const long long*)d_in[1]; // int64 [2, E]; row 0 = target
    const float*     node_attr  = (const float*)d_in[2];
    const float*     W_w        = (const float*)d_in[3];     // [512, 640]
    const float*     W_b        = (const float*)d_in[4];     // [512]
    float*           out        = (float*)d_out;

    // Workspace layout
    float* sums   = (float*)d_ws;                              // N * 128 f32  (51.2 MB)
    float* counts = sums + (size_t)N_NODES * EDGE_FEAT;        // N f32        (0.4 MB)
    unsigned short* Wb = (unsigned short*)(counts + N_NODES);  // 512*640 bf16 (0.66 MB)
    unsigned short* hbf = Wb + (size_t)OUT_CH * K_DIM;         // N*640 bf16   (128 MB)

    // 1) zero sums + counts (contiguous)
    {
        int n = N_NODES * EDGE_FEAT + N_NODES;
        zero_f32_kernel<<<(n + 255) / 256, 256, 0, stream>>>(sums, n);
    }
    // 2) convert W to bf16
    {
        int n = OUT_CH * K_DIM;
        convw_kernel<<<(n + 255) / 256, 256, 0, stream>>>(W_w, Wb, n);
    }
    // 3) scatter-sum (f32 hardware atomics)
    {
        long long work = (long long)N_EDGES * 32;
        scatter_kernel<<<(int)((work + 255) / 256), 256, 0, stream>>>(
            edge_attr, edge_index, sums, counts);
    }
    // 4) pack h = [node_attr | mean] as bf16 (single pass over node_attr)
    {
        long long work = (long long)N_NODES * (K_DIM / 8);
        pack_h_kernel<<<(int)((work + 255) / 256), 256, 0, stream>>>(
            node_attr, sums, counts, hbf);
    }
    // 5) bf16 WMMA GEMM + bias + sigmoid (double-buffered async LDS staging)
    {
        dim3 grid(OUT_CH / BN, (N_NODES + BM - 1) / BM); // col-major: L2 reuse of h
        size_t ldsBytes = 2u * BN * KC * sizeof(unsigned short); // 128 KB
        gemm_kernel<<<grid, 256, ldsBytes, stream>>>(hbf, Wb, W_b, out);
    }
}
